// GAT_4037269259015
// MI455X (gfx1250) — compile-verified, matrix operand order checked by reading, not run on hardware
//
#include <hip/hip_runtime.h>
#include <hip/hip_bf16.h>
#include <math.h>

typedef __attribute__((ext_vector_type(16))) _Float16 v16h;
typedef __attribute__((ext_vector_type(8)))  _Float16 v8h;
typedef __attribute__((ext_vector_type(8)))  float    v8f;

static __device__ __forceinline__ float leaky02(float v) {
    return v > 0.f ? v : 0.2f * v;
}

// float atomic max via sign-aware integer atomics (works with -inf init)
static __device__ __forceinline__ void atomicMaxFloat(float* addr, float val) {
    if (val >= 0.f) {
        atomicMax((int*)addr, __float_as_int(val));
    } else {
        atomicMin((unsigned int*)addr, __float_as_uint(val));
    }
}

// ---------------- utility kernels ----------------
__global__ void k_fill(float* p, float v, long long n) {
    long long t = (long long)blockIdx.x * blockDim.x + threadIdx.x;
    if (t < n) p[t] = v;
}

__global__ void k_f32_to_f16(const float* __restrict__ in, _Float16* __restrict__ out, long long n) {
    long long t = (long long)blockIdx.x * blockDim.x + threadIdx.x;
    if (t < n) out[t] = (_Float16)in[t];
}

// W is [K, Nout] row-major f32 -> WT [n, k] f16 (row stride K)
__global__ void k_transpose_w(const float* __restrict__ W, _Float16* __restrict__ WT, int K, int Nout) {
    int t = blockIdx.x * blockDim.x + threadIdx.x;
    if (t >= K * Nout) return;
    int k = t / Nout, n = t % Nout;
    WT[(size_t)n * K + k] = (_Float16)W[t];
}

// ---------------- WMMA GEMM: C[M,Nout] = A[M,K] * BT[Npad,K]^T ----------------
// Each wave computes a 16 x (16*NT) tile: A registers are loaded once per K-step
// and reused across NT WMMAs (4x less A traffic for NT=4).
template <int NT>
__global__ __launch_bounds__(256)
void k_gemm_wmma(const _Float16* __restrict__ A,
                 const _Float16* __restrict__ BT,
                 float* __restrict__ C,
                 int M, int K, int Nout, int Npad)
{
    const int ngroups = Npad / (16 * NT);
    const int wave = blockIdx.x * (blockDim.x >> 5) + (threadIdx.x >> 5);
    const int mt = wave / ngroups;
    const int ng = wave % ngroups;
    if (mt * 16 >= M) return;                 // wave-uniform guard (EXEC stays all-1 for WMMA)

    const int lane = threadIdx.x & 31;
    const int half = lane >> 4;
    const int idx  = lane & 15;

    const _Float16* arow = A  + (size_t)(mt * 16 + idx) * K;
    const _Float16* brow = BT + (size_t)(ng * NT * 16 + idx) * K;   // + j*16*K per n-tile

    v8f acc[NT];
    #pragma unroll
    for (int j = 0; j < NT; ++j) acc[j] = (v8f){};

    for (int k0 = 0; k0 < K; k0 += 32) {
        // A 16x32: lane holds row idx; elems 0..7 -> K=k0+half*8+j, elems 8..15 -> K=k0+16+half*8+j
        v8h alo = *(const v8h*)(arow + k0 + half * 8);
        v8h ahi = *(const v8h*)(arow + k0 + 16 + half * 8);
        v16h av = __builtin_shufflevector(alo, ahi, 0,1,2,3,4,5,6,7,8,9,10,11,12,13,14,15);
        #pragma unroll
        for (int j = 0; j < NT; ++j) {
            // B 32x16 mirrors A with N in place of M (BT row = B column)
            const _Float16* bj = brow + (size_t)j * 16 * K;
            v8h blo = *(const v8h*)(bj + k0 + half * 8);
            v8h bhi = *(const v8h*)(bj + k0 + 16 + half * 8);
            v16h bv = __builtin_shufflevector(blo, bhi, 0,1,2,3,4,5,6,7,8,9,10,11,12,13,14,15);
            acc[j] = __builtin_amdgcn_wmma_f32_16x16x32_f16(false, av, false, bv,
                                                            (short)0, acc[j], false, false);
        }
    }

    #pragma unroll
    for (int j = 0; j < NT; ++j) {
        const int ncol = (ng * NT + j) * 16 + idx;
        if (ncol < Nout) {
            #pragma unroll
            for (int v = 0; v < 8; ++v) {
                int mrow = mt * 16 + v + half * 8;   // C: VGPR v -> M = v + 8*half, N = idx
                C[(size_t)mrow * Nout + ncol] = acc[j][v];
            }
        }
    }
}

// ---------------- attention kernels ----------------
// el[n,h] = sum_d H[n, h*D+d]*al[h,d] ; er likewise
__global__ void k_el_er(const float* __restrict__ Hm, const float* __restrict__ al,
                        const float* __restrict__ ar, float* __restrict__ el,
                        float* __restrict__ er, int N, int H, int D)
{
    int t = blockIdx.x * blockDim.x + threadIdx.x;
    if (t >= N * H) return;
    int n = t / H, h = t % H;
    const float* row = Hm + (size_t)n * H * D + (size_t)h * D;
    float a = 0.f, b = 0.f;
    for (int d = 0; d < D; ++d) {
        float x = row[d];
        a += x * al[h * D + d];
        b += x * ar[h * D + d];
    }
    el[t] = a; er[t] = b;
}

__global__ void k_edge_max(const int* __restrict__ src, const int* __restrict__ dst,
                           const float* __restrict__ el, const float* __restrict__ er,
                           float* __restrict__ mx, int E, int H)
{
    int t = blockIdx.x * blockDim.x + threadIdx.x;
    if (t >= E * H) return;
    int e = t / H, h = t % H;
    int s = src[e], d = dst[e];
    float v = leaky02(el[s * H + h] + er[d * H + h]);
    atomicMaxFloat(&mx[d * H + h], v);
}

// computes ex = exp(e - m[dst]); accumulates segment sum AND stores ex per (edge,head)
__global__ void k_edge_sum(const int* __restrict__ src, const int* __restrict__ dst,
                           const float* __restrict__ el, const float* __restrict__ er,
                           const float* __restrict__ mx, float* __restrict__ sm,
                           float* __restrict__ exb, int E, int H)
{
    int t = blockIdx.x * blockDim.x + threadIdx.x;
    if (t >= E * H) return;
    int e = t / H, h = t % H;
    int s = src[e], d = dst[e];
    float v = leaky02(el[s * H + h] + er[d * H + h]);
    float ex = __expf(v - mx[d * H + h]);
    exb[t] = ex;
    atomicAdd(&sm[d * H + h], ex);
}

// one thread per (edge, feature): coalesced gather of h[src] row, coalesced atomics to out[dst] row
__global__ void k_edge_agg(const int* __restrict__ src, const int* __restrict__ dst,
                           const float* __restrict__ Hm, const float* __restrict__ exb,
                           const float* __restrict__ sm, float* __restrict__ out,
                           int E, int H, int D)
{
    const int F = H * D;
    long long t = (long long)blockIdx.x * blockDim.x + threadIdx.x;
    if (t >= (long long)E * F) return;
    int e = (int)(t / F), f = (int)(t % F);
    int h = f / D;
    int s = src[e], d = dst[e];
    float a = exb[e * H + h] / sm[d * H + h];
    atomicAdd(&out[(size_t)d * F + f], Hm[(size_t)s * F + f] * a);
}

// out = elu(out + bias); also emit f16 copy for next layer's GEMM
__global__ void k_finish(float* __restrict__ out, const float* __restrict__ bias,
                         _Float16* __restrict__ xh, long long total, int F)
{
    long long t = (long long)blockIdx.x * blockDim.x + threadIdx.x;
    if (t >= total) return;
    int f = (int)(t % F);
    float v = out[t] + bias[f];
    v = v > 0.f ? v : __expf(v) - 1.f;
    out[t] = v;
    xh[t] = (_Float16)v;
}

__global__ void k_bias_add(float* __restrict__ out, const float* __restrict__ bias,
                           long long total, int F)
{
    long long t = (long long)blockIdx.x * blockDim.x + threadIdx.x;
    if (t >= total) return;
    out[t] += bias[(int)(t % F)];
}

// ---------------- host side ----------------
static inline unsigned cdivll(long long a, int b) { return (unsigned)((a + b - 1) / b); }

extern "C" void kernel_launch(void* const* d_in, const int* in_sizes, int n_in,
                              void* d_out, int out_size, void* d_ws, size_t ws_size,
                              hipStream_t stream) {
    const float* feat = (const float*)d_in[0];
    const int*   src  = (const int*)d_in[1];
    const int*   dst  = (const int*)d_in[2];
    const float* W1  = (const float*)d_in[3];
    const float* al1 = (const float*)d_in[4];
    const float* ar1 = (const float*)d_in[5];
    const float* b1  = (const float*)d_in[6];
    const float* W2  = (const float*)d_in[7];
    const float* al2 = (const float*)d_in[8];
    const float* ar2 = (const float*)d_in[9];
    const float* b2  = (const float*)d_in[10];
    const float* W3  = (const float*)d_in[11];
    const float* al3 = (const float*)d_in[12];
    const float* ar3 = (const float*)d_in[13];
    const float* b3  = (const float*)d_in[14];

    const int N = in_sizes[0] / 128;   // 50000
    const int E = in_sizes[1];         // 800000
    const int blk = 256;

    // workspace carve
    auto au = [](size_t x) { return (x + 255) & ~(size_t)255; };
    char* w = (char*)d_ws;
    _Float16* xh = (_Float16*)w; w += au((size_t)N * 256 * sizeof(_Float16));
    _Float16* wT = (_Float16*)w; w += au((size_t)256 * 256 * sizeof(_Float16));
    float* Hb  = (float*)w; w += au((size_t)N * 256 * sizeof(float));
    float* Ob  = (float*)w; w += au((size_t)N * 256 * sizeof(float));
    float* elb = (float*)w; w += au((size_t)N * 4 * sizeof(float));
    float* erb = (float*)w; w += au((size_t)N * 4 * sizeof(float));
    float* mxb = (float*)w; w += au((size_t)N * 4 * sizeof(float));
    float* smb = (float*)w; w += au((size_t)N * 4 * sizeof(float));
    float* exb = (float*)w; w += au((size_t)E * 4 * sizeof(float));
    float* outp = (float*)d_out;

    const float NEG_INF = -INFINITY;

    // ================= Layer 1: F_in=128, F_out=256, H=4, D=64 =================
    k_f32_to_f16<<<cdivll((long long)N * 128, blk), blk, 0, stream>>>(feat, xh, (long long)N * 128);
    k_transpose_w<<<cdivll(128 * 256, blk), blk, 0, stream>>>(W1, wT, 128, 256);
    {
        long long waves = (long long)(N / 16) * 4;   // ngroups = 256/64
        k_gemm_wmma<4><<<cdivll(waves, 8), 256, 0, stream>>>(xh, wT, Hb, N, 128, 256, 256);
    }
    k_el_er<<<cdivll((long long)N * 4, blk), blk, 0, stream>>>(Hb, al1, ar1, elb, erb, N, 4, 64);
    k_fill<<<cdivll((long long)N * 4, blk), blk, 0, stream>>>(mxb, NEG_INF, (long long)N * 4);
    k_fill<<<cdivll((long long)N * 4, blk), blk, 0, stream>>>(smb, 0.f, (long long)N * 4);
    k_fill<<<cdivll((long long)N * 256, blk), blk, 0, stream>>>(Ob, 0.f, (long long)N * 256);
    k_edge_max<<<cdivll((long long)E * 4, blk), blk, 0, stream>>>(src, dst, elb, erb, mxb, E, 4);
    k_edge_sum<<<cdivll((long long)E * 4, blk), blk, 0, stream>>>(src, dst, elb, erb, mxb, smb, exb, E, 4);
    k_edge_agg<<<cdivll((long long)E * 256, blk), blk, 0, stream>>>(src, dst, Hb, exb, smb, Ob, E, 4, 64);
    k_finish<<<cdivll((long long)N * 256, blk), blk, 0, stream>>>(Ob, b1, xh, (long long)N * 256, 256);

    // ================= Layer 2: F_in=256, F_out=256, H=4, D=64 =================
    k_transpose_w<<<cdivll(256 * 256, blk), blk, 0, stream>>>(W2, wT, 256, 256);
    {
        long long waves = (long long)(N / 16) * 4;
        k_gemm_wmma<4><<<cdivll(waves, 8), 256, 0, stream>>>(xh, wT, Hb, N, 256, 256, 256);
    }
    k_el_er<<<cdivll((long long)N * 4, blk), blk, 0, stream>>>(Hb, al2, ar2, elb, erb, N, 4, 64);
    k_fill<<<cdivll((long long)N * 4, blk), blk, 0, stream>>>(mxb, NEG_INF, (long long)N * 4);
    k_fill<<<cdivll((long long)N * 4, blk), blk, 0, stream>>>(smb, 0.f, (long long)N * 4);
    k_fill<<<cdivll((long long)N * 256, blk), blk, 0, stream>>>(Ob, 0.f, (long long)N * 256);
    k_edge_max<<<cdivll((long long)E * 4, blk), blk, 0, stream>>>(src, dst, elb, erb, mxb, E, 4);
    k_edge_sum<<<cdivll((long long)E * 4, blk), blk, 0, stream>>>(src, dst, elb, erb, mxb, smb, exb, E, 4);
    k_edge_agg<<<cdivll((long long)E * 256, blk), blk, 0, stream>>>(src, dst, Hb, exb, smb, Ob, E, 4, 64);
    k_finish<<<cdivll((long long)N * 256, blk), blk, 0, stream>>>(Ob, b2, xh, (long long)N * 256, 256);

    // ================= Layer 3: F_in=256, F_out=8 (padded to 16), H=1, D=8 =================
    k_fill<<<cdivll(2048, blk), blk, 0, stream>>>((float*)wT, 0.f, 2048); // zero 16x256 f16 region
    k_transpose_w<<<cdivll(256 * 8, blk), blk, 0, stream>>>(W3, wT, 256, 8);
    {
        long long waves = (long long)(N / 16);       // ngroups = 16/16 = 1
        k_gemm_wmma<1><<<cdivll(waves, 8), 256, 0, stream>>>(xh, wT, Hb, N, 256, 8, 16);
    }
    k_el_er<<<cdivll((long long)N, blk), blk, 0, stream>>>(Hb, al3, ar3, elb, erb, N, 1, 8);
    k_fill<<<cdivll((long long)N, blk), blk, 0, stream>>>(mxb, NEG_INF, (long long)N);
    k_fill<<<cdivll((long long)N, blk), blk, 0, stream>>>(smb, 0.f, (long long)N);
    k_fill<<<cdivll((long long)N * 8, blk), blk, 0, stream>>>(outp, 0.f, (long long)N * 8);
    k_edge_max<<<cdivll((long long)E, blk), blk, 0, stream>>>(src, dst, elb, erb, mxb, E, 1);
    k_edge_sum<<<cdivll((long long)E, blk), blk, 0, stream>>>(src, dst, elb, erb, mxb, smb, exb, E, 1);
    k_edge_agg<<<cdivll((long long)E * 8, blk), blk, 0, stream>>>(src, dst, Hb, exb, smb, outp, E, 1, 8);
    // head mean over H=1 is identity; just add bias, no activation
    k_bias_add<<<cdivll((long long)N * 8, blk), blk, 0, stream>>>(outp, b3, (long long)N * 8, 8);
}
